// BRITS_85813446574884
// MI455X (gfx1250) — matrix operand order repaired and checked by reading, not compiled, not run on hardware
//
#include <hip/hip_runtime.h>
#include <hip/hip_bf16.h>
#include <stdint.h>

// ---------------------------------------------------------------------------
// BRITS on MI455X (gfx1250): persistent per-batch-tile scan kernel using
// v_wmma_f32_16x16x32_bf16 for every GEMM. 32 workgroups (16 batch tiles x 2
// directions), h/c state kept in LDS. Weights pre-packed (global, bf16,
// B-operand lane layout, 2x b128/lane/tile). Activations packed into LDS in
// the A-operand lane layout (bank-conflict-free b64 reads); GEMM epilogues
// store their own bf16 outputs directly via 16-bit LDS stores (each lane owns
// one half of a packed dword), so each step needs only 6 workgroup barriers
// and no exec divergence or cross-lane shuffles.
// ---------------------------------------------------------------------------

typedef __attribute__((ext_vector_type(16))) __bf16 v16bf;
typedef __attribute__((ext_vector_type(8)))  float  v8f;

#define B_SZ 256
#define T_SZ 256
#define F_SZ 128
#define H_SZ 256

// per-direction packed-weight dword offsets in workspace
#define O_WDH   0        // 256x128 -> 16384 dwords
#define O_WHIST 16384    // 128x256 -> 16384
#define O_WF    32768    // 128x128 -> 8192
#define O_WCOMB 40960    // 128x256 -> 16384
#define O_WIH   57344    // 1024x256 -> 131072
#define O_WHH   188416   // 1024x256 -> 131072
#define DIR_DW  319488   // total dwords per direction

__device__ __forceinline__ uint16_t f2bf(float f) {
  uint32_t x = __float_as_uint(f);
  uint32_t r = (x + 0x7FFFu + ((x >> 16) & 1u)) >> 16;   // round-to-nearest-even
  return (uint16_t)r;
}
__device__ __forceinline__ uint32_t bfpair(float a, float b) {
  return (uint32_t)f2bf(a) | ((uint32_t)f2bf(b) << 16);
}
__device__ __forceinline__ float sigf(float x) { return 1.0f / (1.0f + __expf(-x)); }

// ---- packed A-operand LDS layout ------------------------------------------
// v16bf element e of the A fragment covers K = 16*(e>>3) + 8*(lane>>4) + (e&7).
// Dword v (elements 2v,2v+1) of lane L is stored at
//   ktile*256 + (v>>1)*64 + L*2 + (v&1)
// so a lane's fragment is 4x ds_load_b64 and the 64 dwords touched per v-pair
// cover all 64 LDS banks exactly once (conflict-free).
__device__ __forceinline__ int pk_idx(int m, int k /*even global K*/) {
  int ktile = k >> 5, kk = k & 31;
  int h = (kk >> 3) & 1;
  int v = ((kk >> 4) << 2) + ((kk & 7) >> 1);
  return (ktile << 8) + ((v >> 1) << 6) + ((h * 16 + m) << 1) + (v & 1);
}

// Per-lane bf16 element store into a packed A-operand buffer (k any parity).
__device__ __forceinline__ void pk_store_half(uint32_t* pk, int m, int k, float val) {
  int dw = pk_idx(m, k & ~1);
  ((uint16_t*)pk)[(dw << 1) | (k & 1)] = f2bf(val);
}

__device__ __forceinline__ v16bf a_frag_pk(const uint32_t* pk, int ktile) {
  int lane = threadIdx.x & 31;
  const uint32_t* p = pk + (ktile << 8) + (lane << 1);
  union { uint32_t u[8]; v16bf v; } a;
#pragma unroll
  for (int v2 = 0; v2 < 4; ++v2) {
    a.u[2 * v2]     = p[v2 * 64];
    a.u[2 * v2 + 1] = p[v2 * 64 + 1];
  }
  return a.v;
}

// Pre-packed 32x16 bf16 B tile in global: 256 dwords, lane's 8 dwords contiguous.
__device__ __forceinline__ v16bf b_frag(const uint32_t* pk, int tile) {
  int lane = threadIdx.x & 31;
  union { uint32_t u[8]; v16bf v; } b;
  const uint32_t* p = pk + ((size_t)tile << 8) + (lane << 3);
#pragma unroll
  for (int i = 0; i < 8; ++i) b.u[i] = p[i];
  return b.v;
}

__device__ __forceinline__ v8f wmma_bf16(v16bf a, v16bf b, v8f c) {
  return __builtin_amdgcn_wmma_f32_16x16x32_bf16(false, a, false, b, (short)0, c, false, false);
}

// ---------------------------------------------------------------------------
// Weight packer: f32 (N rows x K cols, row-major) -> bf16 WMMA-B tiles.
// Tile (nt,kt): B[kk][nn] = W[nt*16+nn][kt*32+kk]; dword idx = lane*8 + v
// holds K pair (16*(v>>2) + 8*(lane>>4) + 2*(v&3), +1) for column nn = lane&15.
// ---------------------------------------------------------------------------
__global__ void brits_pack_w(const float* __restrict__ src, uint32_t* __restrict__ dst,
                             int N, int K, int zero_diag) {
  int g = blockIdx.x * blockDim.x + threadIdx.x;
  int total = (N * K) >> 1;
  if (g >= total) return;
  int numK = K >> 5;
  int tile = g >> 8, r = g & 255;
  int lane = r >> 3, v = r & 7;
  int nt = tile / numK, kt = tile - nt * numK;
  int n = nt * 16 + (lane & 15);
  int k = kt * 32 + ((v >> 2) << 4) + ((lane >> 4) << 3) + ((v & 3) << 1);
  float a0 = src[(size_t)n * K + k];
  float a1 = src[(size_t)n * K + k + 1];
  if (zero_diag) { if (n == k) a0 = 0.0f; if (n == k + 1) a1 = 0.0f; }
  dst[g] = bfpair(a0, a1);
}

__global__ void brits_init_out(const float* __restrict__ X, const float* __restrict__ M,
                               float* __restrict__ out, int n) {
  int i = blockIdx.x * blockDim.x + threadIdx.x;
  if (i < n) out[i] = M[i] * X[i];
}

// ---------------------------------------------------------------------------
struct RitsArgs {
  const float *X, *M, *Dlt;
  const uint32_t *pWdh, *pWhist, *pWf, *pWcomb, *pWih, *pWhh;
  const float *Wdx, *bdh, *bdx, *bhist, *bfeat, *bcomb, *bih, *bhh;
  float* out;
  int reversed;
};
struct ArgsPair { RitsArgs fw, bw; };

__global__ __launch_bounds__(256, 1)
void brits_scan(ArgsPair ap) {
  // f32 tiles (padded: stride mod 64 == 4)
  __shared__ float sh_h[16][H_SZ + 4];        // hidden state (persistent)
  __shared__ float sh_c[16][H_SZ + 4];        // cell state   (persistent)
  __shared__ float sh_x[16][F_SZ + 4];
  __shared__ float sh_m[16][F_SZ + 4];
  __shared__ float sh_g[16][4 * H_SZ + 4];    // LSTM gates
  // bf16 A-operand tiles in packed lane layout
  __shared__ uint32_t pk_d[4 * 256];          // deltas, then x_c   (K=128)
  __shared__ uint32_t pk_inp[8 * 256];        // [gamma_x|m] -> [c_c|m] (K=256)
  __shared__ uint32_t pk_h[8 * 256];          // decayed h          (K=256)

  const RitsArgs& A = (blockIdx.x < 16) ? ap.fw : ap.bw;
  const int bt   = (blockIdx.x & 15) << 4;    // batch row base (16 rows)
  const int tid  = threadIdx.x;
  const int lane = tid & 31;
  const int w    = tid >> 5;                  // wave id, 8 waves (wave32)
  const int nn   = lane & 15;
  const int hh   = lane >> 4;

  for (int i = tid; i < 16 * (H_SZ + 4); i += 256) {
    (&sh_h[0][0])[i] = 0.0f;
    (&sh_c[0][0])[i] = 0.0f;
  }
  __syncthreads();

  for (int t = 0; t < T_SZ; ++t) {
    const int tg = A.reversed ? (T_SZ - 1 - t) : t;   // time index into X/M/out

    // ---- ph0: load x,m,d pairs; pack deltas, gamma_x, m --------------------
    for (int i = tid; i < 16 * 64; i += 256) {        // 1024 float2 pairs
      int r = i >> 6, fp = (i & 63) << 1;
      size_t gidx = ((size_t)(bt + r) * T_SZ + tg) * F_SZ + fp;
      size_t didx = ((size_t)(bt + r) * T_SZ + t) * F_SZ + fp;
      float2 xv = *(const float2*)&A.X[gidx];
      float2 mv = *(const float2*)&A.M[gidx];
      float2 dv = *(const float2*)&A.Dlt[didx];
      sh_x[r][fp] = xv.x; sh_x[r][fp + 1] = xv.y;
      sh_m[r][fp] = mv.x; sh_m[r][fp + 1] = mv.y;
      pk_d[pk_idx(r, fp)] = bfpair(dv.x, dv.y);
      float g0 = __expf(-fmaxf(dv.x * A.Wdx[(size_t)fp * F_SZ + fp] + A.bdx[fp], 0.0f));
      float g1 = __expf(-fmaxf(dv.y * A.Wdx[(size_t)(fp + 1) * F_SZ + fp + 1] + A.bdx[fp + 1], 0.0f));
      pk_inp[pk_idx(r, fp)] = bfpair(g0, g1);
      pk_inp[pk_idx(r, F_SZ + fp)] = bfpair(mv.x, mv.y);
      if (t + 1 < T_SZ) {
        int tn = A.reversed ? (T_SZ - 2 - t) : (t + 1);
        __builtin_prefetch(&A.X[((size_t)(bt + r) * T_SZ + tn) * F_SZ + fp], 0, 0);
      }
    }
    __syncthreads();

    // ---- ph1: gamma_h = exp(-relu(D*Wdh^T + bdh)); h *= gamma_h; pack h ----
    {
      v16bf ad[4];
#pragma unroll
      for (int kt = 0; kt < 4; ++kt) ad[kt] = a_frag_pk(pk_d, kt);
#pragma unroll
      for (int nti = 0; nti < 2; ++nti) {
        int nt = w * 2 + nti;                         // 16 n-tiles, 2/wave
        v8f acc = {};
#pragma unroll
        for (int kt = 0; kt < 4; ++kt)
          acc = wmma_bf16(ad[kt], b_frag(A.pWdh, nt * 4 + kt), acc);
        int n = nt * 16 + nn;
        float bias = A.bdh[n];
#pragma unroll
        for (int r = 0; r < 8; ++r) {
          int m = r + (hh << 3);
          float hv = sh_h[m][n] * __expf(-fmaxf(acc[r] + bias, 0.0f));
          sh_h[m][n] = hv;
          pk_store_half(pk_h, m, n, hv);              // 16-bit LDS store, all lanes
        }
      }
    }
    __syncthreads();

    // ---- ph2: alpha (regs), x_h (regs), x_c packed into pk_d ---------------
    float al[8], xh[8], mvr[8], xvr[8];
    {
      int nt = w;                                     // 8 n-tiles, 1/wave
      int n = nt * 16 + nn;
      v8f acca = {};
#pragma unroll
      for (int kt = 0; kt < 8; ++kt)
        acca = wmma_bf16(a_frag_pk(pk_inp, kt), b_frag(A.pWcomb, nt * 8 + kt), acca);
      v8f acch = {};
#pragma unroll
      for (int kt = 0; kt < 8; ++kt)
        acch = wmma_bf16(a_frag_pk(pk_h, kt), b_frag(A.pWhist, nt * 8 + kt), acch);
      float ba = A.bcomb[n], bh = A.bhist[n];
#pragma unroll
      for (int r = 0; r < 8; ++r) {
        int m = r + (hh << 3);
        al[r] = sigf(acca[r] + ba);
        float x_h = acch[r] + bh;
        xh[r] = x_h;
        float mv = sh_m[m][n]; mvr[r] = mv;
        float xv = sh_x[m][n]; xvr[r] = xv;
        float xc = mv * xv + (1.0f - mv) * x_h;
        pk_store_half(pk_d, m, n, xc);
      }
    }
    __syncthreads();

    // ---- ph3: z_h = x_c*Wf^T + bfeat; c_h, c_c packed; emit estimation -----
    {
      int nt = w;
      int n = nt * 16 + nn;
      v8f acc = {};
#pragma unroll
      for (int kt = 0; kt < 4; ++kt)
        acc = wmma_bf16(a_frag_pk(pk_d, kt), b_frag(A.pWf, nt * 4 + kt), acc);
      float bias = A.bfeat[n];
#pragma unroll
      for (int r = 0; r < 8; ++r) {
        int m = r + (hh << 3);
        float z  = acc[r] + bias;
        float ch = al[r] * z + (1.0f - al[r]) * xh[r];
        float cc = mvr[r] * xvr[r] + (1.0f - mvr[r]) * ch;
        pk_store_half(pk_inp, m, n, cc);
        atomicAdd(&A.out[((size_t)(bt + m) * T_SZ + tg) * F_SZ + n],
                  0.5f * (1.0f - mvr[r]) * ch);
      }
    }
    __syncthreads();

    // ---- ph4: gates = [c_c,m]*Wih^T + h*Whh^T + bih + bhh ------------------
    {
      v16bf ai[8], ah[8];
#pragma unroll
      for (int kt = 0; kt < 8; ++kt) {
        ai[kt] = a_frag_pk(pk_inp, kt);
        ah[kt] = a_frag_pk(pk_h, kt);
      }
#pragma unroll
      for (int nti = 0; nti < 8; ++nti) {
        int nt = w * 8 + nti;                         // 64 n-tiles, 8/wave
        v8f acc = {};
#pragma unroll
        for (int kt = 0; kt < 8; ++kt)
          acc = wmma_bf16(ai[kt], b_frag(A.pWih, nt * 8 + kt), acc);
#pragma unroll
        for (int kt = 0; kt < 8; ++kt)
          acc = wmma_bf16(ah[kt], b_frag(A.pWhh, nt * 8 + kt), acc);
        int n = nt * 16 + nn;
        float bias = A.bih[n] + A.bhh[n];
#pragma unroll
        for (int r = 0; r < 8; ++r)
          sh_g[r + (hh << 3)][n] = acc[r] + bias;
      }
    }
    __syncthreads();

    // ---- ph5: LSTM state update (torch gate order i,f,g,o) -----------------
    for (int i = tid; i < 16 * H_SZ; i += 256) {
      int r = i >> 8, j = i & 255;
      float ig = sigf(sh_g[r][j]);
      float fg = sigf(sh_g[r][H_SZ + j]);
      float gg = tanhf(sh_g[r][2 * H_SZ + j]);
      float og = sigf(sh_g[r][3 * H_SZ + j]);
      float cv = fg * sh_c[r][j] + ig * gg;
      sh_c[r][j] = cv;
      sh_h[r][j] = og * tanhf(cv);
    }
    __syncthreads();
  }
}

// ---------------------------------------------------------------------------
extern "C" void kernel_launch(void* const* d_in, const int* in_sizes, int n_in,
                              void* d_out, int out_size, void* d_ws, size_t ws_size,
                              hipStream_t stream) {
  (void)in_sizes; (void)n_in; (void)ws_size;
  const float* X  = (const float*)d_in[0];
  const float* M  = (const float*)d_in[1];
  const float* Df = (const float*)d_in[2];
  const float* Db = (const float*)d_in[3];
  uint32_t* ws = (uint32_t*)d_ws;

  // pack weights (bf16, WMMA-B layout) for both directions
  for (int dir = 0; dir < 2; ++dir) {
    int off = 4 + dir * 14;                       // param dict order per direction
    uint32_t* base = ws + (size_t)dir * DIR_DW;
    struct { int idx, o, N, K, zd; } jobs[6] = {
      { off + 0,  O_WDH,   256,  128, 0 },        // Wdh
      { off + 4,  O_WHIST, 128,  256, 0 },        // Whist
      { off + 6,  O_WF,    128,  128, 1 },        // Wfeat (zero diagonal)
      { off + 8,  O_WCOMB, 128,  256, 0 },        // Wcomb
      { off + 10, O_WIH,   1024, 256, 0 },        // Wih
      { off + 12, O_WHH,   1024, 256, 0 },        // Whh
    };
    for (int j = 0; j < 6; ++j) {
      int total = (jobs[j].N * jobs[j].K) >> 1;
      brits_pack_w<<<(total + 255) / 256, 256, 0, stream>>>(
          (const float*)d_in[jobs[j].idx], base + jobs[j].o,
          jobs[j].N, jobs[j].K, jobs[j].zd);
    }
  }

  // out = M*X, then each direction atomically adds 0.5*(1-M)*est
  brits_init_out<<<(out_size + 255) / 256, 256, 0, stream>>>(X, M, (float*)d_out, out_size);

  ArgsPair ap;
  for (int dir = 0; dir < 2; ++dir) {
    int off = 4 + dir * 14;
    uint32_t* base = ws + (size_t)dir * DIR_DW;
    RitsArgs a;
    a.X = X; a.M = M;
    a.Dlt   = dir ? Db : Df;
    a.pWdh  = base + O_WDH;   a.pWhist = base + O_WHIST;
    a.pWf   = base + O_WF;    a.pWcomb = base + O_WCOMB;
    a.pWih  = base + O_WIH;   a.pWhh   = base + O_WHH;
    a.Wdx   = (const float*)d_in[off + 2];
    a.bdh   = (const float*)d_in[off + 1];
    a.bdx   = (const float*)d_in[off + 3];
    a.bhist = (const float*)d_in[off + 5];
    a.bfeat = (const float*)d_in[off + 7];
    a.bcomb = (const float*)d_in[off + 9];
    a.bih   = (const float*)d_in[off + 11];
    a.bhh   = (const float*)d_in[off + 13];
    a.out   = (float*)d_out;
    a.reversed = dir;
    if (dir == 0) ap.fw = a; else ap.bw = a;
  }
  brits_scan<<<32, 256, 0, stream>>>(ap);
}
